// GraphConv_16363825397897
// MI455X (gfx1250) — matrix operand order repaired and checked by reading, not compiled, not run on hardware
//
#include <hip/hip_runtime.h>

// GCN layer: out = relu( D^-1/2 (A+I) D^-1/2 @ H @ W )
// N = 8192, D_IN = D_OUT = 512. bf16 WMMA path for both GEMMs.
// GEMM kernels software-pipeline global->reg staging against WMMA compute.

#define GN 8192
#define GD 512

#define BM 128
#define BN 128
#define BK 64
#define BKP 72   // LDS row pitch (elements): 144 B, conflict-free & 16B-aligned

typedef __attribute__((ext_vector_type(16))) __bf16 v16bf;
typedef __attribute__((ext_vector_type(8)))  __bf16 v8bf;
typedef __attribute__((ext_vector_type(4)))  __bf16 v4bf;
typedef __attribute__((ext_vector_type(2)))  __bf16 v2bf;
typedef __attribute__((ext_vector_type(8)))  float  v8f;

__device__ __forceinline__ __bf16 f2bf(float f) {
  union { float f; unsigned u; } v; v.f = f;
  unsigned r = v.u + 0x7FFFu + ((v.u >> 16) & 1u);   // round-to-nearest-even
  union { unsigned short s; __bf16 b; } o; o.s = (unsigned short)(r >> 16);
  return o.b;
}

// Packed f32x2 -> bf16x2 via the hardware packed convert (1 VALU / 2 elems).
// dst[15:0] = bf16(a), dst[31:16] = bf16(b).
__device__ __forceinline__ v2bf f2bf2(float a, float b) {
  unsigned out;
  asm("v_cvt_pk_bf16_f32 %0, %1, %2" : "=v"(out) : "v"(a), "v"(b));
  return __builtin_bit_cast(v2bf, out);
}

__device__ __forceinline__ v8bf cat2222(v2bf a, v2bf b, v2bf c, v2bf d) {
  v4bf ab = __builtin_shufflevector(a, b, 0, 1, 2, 3);
  v4bf cd = __builtin_shufflevector(c, d, 0, 1, 2, 3);
  return __builtin_shufflevector(ab, cd, 0, 1, 2, 3, 4, 5, 6, 7);
}

__device__ __forceinline__ v16bf cat8(v8bf lo, v8bf hi) {
  return __builtin_shufflevector(lo, hi, 0,1,2,3,4,5,6,7,8,9,10,11,12,13,14,15);
}

// ---------------------------------------------------------------------------
// Kernel 1: s[i] = rsqrt(1 + sum_j A[i][j])   (one block of 256 per row)
// ---------------------------------------------------------------------------
__global__ __launch_bounds__(256)
void k_row_rsqrt_deg(const float* __restrict__ A, float* __restrict__ s) {
  __shared__ float red[256];
  const int row = blockIdx.x;
  const float4* Arow = reinterpret_cast<const float4*>(A + (size_t)row * GN);
  float acc = 0.f;
  for (int i = threadIdx.x; i < GN / 4; i += 256) {
    float4 v = Arow[i];
    acc += (v.x + v.y) + (v.z + v.w);
  }
  red[threadIdx.x] = acc;
  __syncthreads();
  #pragma unroll
  for (int off = 128; off > 0; off >>= 1) {
    if (threadIdx.x < off) red[threadIdx.x] += red[threadIdx.x + off];
    __syncthreads();
  }
  if (threadIdx.x == 0) s[row] = rsqrtf(red[0] + 1.0f);
}

// ---------------------------------------------------------------------------
// Kernel 2/3: out[d][j] = bf16( in[j][d] * (scale ? scale[j] : 1) )
// ---------------------------------------------------------------------------
__global__ __launch_bounds__(256)
void k_transpose_scale_bf16(const float* __restrict__ in,
                            const float* __restrict__ scale,
                            __bf16* __restrict__ out, int J, int Dd) {
  __shared__ float tile[32][33];
  const int jBase = blockIdx.x * 32;
  const int dBase = blockIdx.y * 32;
  const int tx = threadIdx.x, ty = threadIdx.y;
  #pragma unroll
  for (int r = 0; r < 32; r += 8) {
    int j = jBase + ty + r;
    float sc = scale ? scale[j] : 1.0f;
    tile[ty + r][tx] = in[(size_t)j * Dd + dBase + tx] * sc;
  }
  __syncthreads();
  #pragma unroll
  for (int r = 0; r < 32; r += 8) {
    int d = dBase + ty + r;
    out[(size_t)d * J + jBase + tx] = f2bf(tile[tx][ty + r]);
  }
}

// ---------------------------------------------------------------------------
// Kernel 4: P[i][d] = bf16( s[i] * sum_j (A[i][j] + (i==j)) * HsT[d][j] )
// Pipeline per k-step: LDS stores (staged regs) -> barrier -> prefetch next
// tile global->regs -> 16x WMMA -> barrier.
// ---------------------------------------------------------------------------
__global__ __launch_bounds__(256)
void k_gemm_ah(const float* __restrict__ A, const __bf16* __restrict__ Bt,
               const float* __restrict__ s, __bf16* __restrict__ P) {
  __shared__ __bf16 As[BM][BKP];
  __shared__ __bf16 Bs[BN][BKP];

  const int tid  = threadIdx.x;
  const int wave = tid >> 5;
  const int lane = tid & 31;
  const int m = lane & 15;
  const int h = lane >> 4;

  const int rBase = blockIdx.y * BM;
  const int cBase = blockIdx.x * BN;

  v8f acc[8];
  #pragma unroll
  for (int i = 0; i < 8; ++i)
    #pragma unroll
    for (int j = 0; j < 8; ++j) acc[i][j] = 0.0f;

  // A-tile mapping: thread -> rows (tid>>2) and (tid>>2)+64, 16 floats each
  const int ar = tid >> 2;
  const int ac = (tid & 3) * 16;
  // B-tile mapping: row = tid&127, 32 elements starting at (tid>>7)*32
  const int br = tid & 127;
  const int bc = (tid >> 7) * 32;

  float4 qa[2][4];   // staged A: 2 rows x 16 f32
  v8bf   qb[4];      // staged B: 32 bf16

  auto loadA = [&](int kb) {
    #pragma unroll
    for (int half = 0; half < 2; ++half) {
      const float4* src =
          (const float4*)(A + (size_t)(rBase + ar + half * 64) * GN + kb + ac);
      qa[half][0] = src[0]; qa[half][1] = src[1];
      qa[half][2] = src[2]; qa[half][3] = src[3];
    }
  };
  auto loadB = [&](int kb) {
    const v8bf* src = (const v8bf*)(Bt + (size_t)(cBase + br) * GN + kb + bc);
    qb[0] = src[0]; qb[1] = src[1]; qb[2] = src[2]; qb[3] = src[3];
  };
  auto storeA = [&](int kb) {
    const bool hasDiag = (kb + BK > rBase) && (kb < rBase + BM);
    #pragma unroll
    for (int half = 0; half < 2; ++half) {
      const int r = ar + half * 64;
      float4 q0 = qa[half][0], q1 = qa[half][1];
      float4 q2 = qa[half][2], q3 = qa[half][3];
      if (hasDiag) {
        const int dc = (rBase + r) - (kb + ac);  // diag pos in this 16-chunk
        if (dc >= 0 && dc < 16) {
          q0.x += (dc ==  0) ? 1.f : 0.f; q0.y += (dc ==  1) ? 1.f : 0.f;
          q0.z += (dc ==  2) ? 1.f : 0.f; q0.w += (dc ==  3) ? 1.f : 0.f;
          q1.x += (dc ==  4) ? 1.f : 0.f; q1.y += (dc ==  5) ? 1.f : 0.f;
          q1.z += (dc ==  6) ? 1.f : 0.f; q1.w += (dc ==  7) ? 1.f : 0.f;
          q2.x += (dc ==  8) ? 1.f : 0.f; q2.y += (dc ==  9) ? 1.f : 0.f;
          q2.z += (dc == 10) ? 1.f : 0.f; q2.w += (dc == 11) ? 1.f : 0.f;
          q3.x += (dc == 12) ? 1.f : 0.f; q3.y += (dc == 13) ? 1.f : 0.f;
          q3.z += (dc == 14) ? 1.f : 0.f; q3.w += (dc == 15) ? 1.f : 0.f;
        }
      }
      v8bf lo = cat2222(f2bf2(q0.x, q0.y), f2bf2(q0.z, q0.w),
                        f2bf2(q1.x, q1.y), f2bf2(q1.z, q1.w));
      v8bf hi = cat2222(f2bf2(q2.x, q2.y), f2bf2(q2.z, q2.w),
                        f2bf2(q3.x, q3.y), f2bf2(q3.z, q3.w));
      *(v8bf*)&As[r][ac]     = lo;
      *(v8bf*)&As[r][ac + 8] = hi;
    }
  };
  auto storeB = [&]() {
    *(v8bf*)&Bs[br][bc]      = qb[0];
    *(v8bf*)&Bs[br][bc + 8]  = qb[1];
    *(v8bf*)&Bs[br][bc + 16] = qb[2];
    *(v8bf*)&Bs[br][bc + 24] = qb[3];
  };

  loadA(0); loadB(0);                     // prologue stage
  for (int kb = 0; kb < GN; kb += BK) {
    storeA(kb); storeB();
    __syncthreads();
    if (kb + BK < GN) { loadA(kb + BK); loadB(kb + BK); }  // prefetch next
    #pragma unroll
    for (int ks = 0; ks < BK; ks += 32) {
      const int arow = wave * 16 + m;
      // A frag (16x32): elems 0..7 -> K=h*8+e ; 8..15 -> K=16+h*8+(e-8)
      v16bf afrag = cat8(*(const v8bf*)&As[arow][ks + h * 8],
                         *(const v8bf*)&As[arow][ks + 16 + h * 8]);
      #pragma unroll
      for (int nt = 0; nt < 8; ++nt) {
        const int bcol = nt * 16 + m;
        // B frag (32x16): elems 0..7 -> K=h*8+e ; 8..15 -> K=16+h*8+(e-8), N = m
        v16bf bfrag = cat8(*(const v8bf*)&Bs[bcol][ks + h * 8],
                           *(const v8bf*)&Bs[bcol][ks + 16 + h * 8]);
        acc[nt] = __builtin_amdgcn_wmma_f32_16x16x32_bf16(
            false, afrag, false, bfrag, (short)0, acc[nt], false, false);
      }
    }
    __syncthreads();
  }
  // epilogue: row scale s[i], store bf16 P.  C/D: M = r + 8h, N = m
  #pragma unroll
  for (int r = 0; r < 8; ++r) {
    const int row = rBase + wave * 16 + r + 8 * h;
    const float sr = s[row];
    #pragma unroll
    for (int nt = 0; nt < 8; ++nt) {
      const int col = cBase + nt * 16 + m;
      P[(size_t)row * GD + col] = f2bf(acc[nt][r] * sr);
    }
  }
}

// ---------------------------------------------------------------------------
// Kernel 5: out = relu( P @ W ) via Wt[o][d], bf16 WMMA, f32 output.
// ---------------------------------------------------------------------------
__global__ __launch_bounds__(256)
void k_gemm_pw(const __bf16* __restrict__ P, const __bf16* __restrict__ Wt,
               float* __restrict__ out) {
  __shared__ __bf16 As[BM][BKP];
  __shared__ __bf16 Bs[BN][BKP];

  const int tid  = threadIdx.x;
  const int wave = tid >> 5;
  const int lane = tid & 31;
  const int m = lane & 15;
  const int h = lane >> 4;

  const int rBase = blockIdx.y * BM;
  const int cBase = blockIdx.x * BN;

  v8f acc[8];
  #pragma unroll
  for (int i = 0; i < 8; ++i)
    #pragma unroll
    for (int j = 0; j < 8; ++j) acc[i][j] = 0.0f;

  const int ar = tid >> 1;          // 0..127
  const int ac = (tid & 1) * 32;    // 0 or 32
  const int br = tid & 127;
  const int bc = (tid >> 7) * 32;

  v8bf qa[4], qb[4];

  auto loadA = [&](int kb) {
    const v8bf* src = (const v8bf*)(P + (size_t)(rBase + ar) * GD + kb + ac);
    qa[0] = src[0]; qa[1] = src[1]; qa[2] = src[2]; qa[3] = src[3];
  };
  auto loadB = [&](int kb) {
    const v8bf* src = (const v8bf*)(Wt + (size_t)(cBase + br) * GD + kb + bc);
    qb[0] = src[0]; qb[1] = src[1]; qb[2] = src[2]; qb[3] = src[3];
  };
  auto storeAB = [&]() {
    *(v8bf*)&As[ar][ac]      = qa[0];
    *(v8bf*)&As[ar][ac + 8]  = qa[1];
    *(v8bf*)&As[ar][ac + 16] = qa[2];
    *(v8bf*)&As[ar][ac + 24] = qa[3];
    *(v8bf*)&Bs[br][bc]      = qb[0];
    *(v8bf*)&Bs[br][bc + 8]  = qb[1];
    *(v8bf*)&Bs[br][bc + 16] = qb[2];
    *(v8bf*)&Bs[br][bc + 24] = qb[3];
  };

  loadA(0); loadB(0);
  for (int kb = 0; kb < GD; kb += BK) {
    storeAB();
    __syncthreads();
    if (kb + BK < GD) { loadA(kb + BK); loadB(kb + BK); }
    #pragma unroll
    for (int ks = 0; ks < BK; ks += 32) {
      const int arow = wave * 16 + m;
      v16bf afrag = cat8(*(const v8bf*)&As[arow][ks + h * 8],
                         *(const v8bf*)&As[arow][ks + 16 + h * 8]);
      #pragma unroll
      for (int nt = 0; nt < 8; ++nt) {
        const int bcol = nt * 16 + m;
        v16bf bfrag = cat8(*(const v8bf*)&Bs[bcol][ks + h * 8],
                           *(const v8bf*)&Bs[bcol][ks + 16 + h * 8]);
        acc[nt] = __builtin_amdgcn_wmma_f32_16x16x32_bf16(
            false, afrag, false, bfrag, (short)0, acc[nt], false, false);
      }
    }
    __syncthreads();
  }
  #pragma unroll
  for (int r = 0; r < 8; ++r) {
    const int row = rBase + wave * 16 + r + 8 * h;
    #pragma unroll
    for (int nt = 0; nt < 8; ++nt) {
      const int col = cBase + nt * 16 + m;
      out[(size_t)row * GD + col] = fmaxf(acc[nt][r], 0.0f);
    }
  }
}

// ---------------------------------------------------------------------------
extern "C" void kernel_launch(void* const* d_in, const int* in_sizes, int n_in,
                              void* d_out, int out_size, void* d_ws, size_t ws_size,
                              hipStream_t stream) {
  (void)in_sizes; (void)n_in; (void)out_size; (void)ws_size;
  const float* A = (const float*)d_in[0];
  const float* H = (const float*)d_in[1];
  const float* W = (const float*)d_in[2];
  float* out = (float*)d_out;

  char* ws = (char*)d_ws;
  float*  s   = (float*)ws;                                   // 32 KB
  __bf16* HsT = (__bf16*)(ws + 32768);                        // 8 MB  [GD][GN]
  __bf16* Wt  = (__bf16*)(ws + 32768 + 8388608);              // 512 KB [GD][GD]
  __bf16* P   = (__bf16*)(ws + 32768 + 8388608 + 524288);     // 8 MB  [GN][GD]

  k_row_rsqrt_deg<<<GN, 256, 0, stream>>>(A, s);
  k_transpose_scale_bf16<<<dim3(GN / 32, GD / 32), dim3(32, 8), 0, stream>>>(
      H, s, HsT, GN, GD);
  k_transpose_scale_bf16<<<dim3(GD / 32, GD / 32), dim3(32, 8), 0, stream>>>(
      W, nullptr, Wt, GD, GD);
  k_gemm_ah<<<dim3(GD / BN, GN / BM), 256, 0, stream>>>(A, HsT, s, P);
  k_gemm_pw<<<dim3(GD / BN, GN / BM), 256, 0, stream>>>(P, Wt, out);
}
